// GraphSage_60103772340409
// MI455X (gfx1250) — matrix operand order repaired and checked by reading, not compiled, not run on hardware
//
#include <hip/hip_runtime.h>
#include <hip/hip_bf16.h>
#include <stddef.h>

// ---------------------------------------------------------------------------
// Types for CDNA5 WMMA (wave32).  v_wmma_f32_16x16x32_bf16:
//   D(16x16 f32, 8 VGPRs) = A(16x32 bf16, 8 VGPRs) * B(32x16 bf16, 8 VGPRs) + C
// ---------------------------------------------------------------------------
typedef __attribute__((ext_vector_type(16))) __bf16 v16bf;
typedef __attribute__((ext_vector_type(8)))  __bf16 v8bf;
typedef __attribute__((ext_vector_type(8)))  float  v8f;
typedef __attribute__((ext_vector_type(4)))  float  v4f;

// ---- A-fragment loaders (16x32 tile; lane = row l&15, khalf = l>>4) -------
// Layout (ISA 7.12.2, 16-bit A 16x32): lanes<16 hold K {0..7,16..23},
// lanes>=16 hold K {8..15,24..31}; 2 elements per VGPR, K-major.
__device__ __forceinline__ v16bf load_a_frag(const float* __restrict__ p) {
  // p points at A[row] + k + khalf*8 (fp32, converted to bf16 in-register)
  v4f x0 = *(const v4f*)(p + 0);
  v4f x1 = *(const v4f*)(p + 4);
  v4f x2 = *(const v4f*)(p + 16);
  v4f x3 = *(const v4f*)(p + 20);
  v16bf r;
#pragma unroll
  for (int j = 0; j < 4; ++j) {
    r[j]      = (__bf16)x0[j];
    r[4 + j]  = (__bf16)x1[j];
    r[8 + j]  = (__bf16)x2[j];
    r[12 + j] = (__bf16)x3[j];
  }
  return r;
}

__device__ __forceinline__ v16bf load_a_frag(const __bf16* __restrict__ p) {
  v8bf lo = *(const v8bf*)(p + 0);
  v8bf hi = *(const v8bf*)(p + 16);
  v16bf r;
#pragma unroll
  for (int j = 0; j < 8; ++j) { r[j] = lo[j]; r[8 + j] = hi[j]; }
  return r;
}

// ---- B-fragment loader (32x16 tile from K-contiguous transposed matrix) ---
// BT is [N][K] row-major bf16.  lane = col l&15; lanes<16 hold K {0..7,16..23},
// lanes>=16 hold K {8..15,24..31}, 2 per VGPR K-major.
__device__ __forceinline__ v16bf load_b_frag(const __bf16* __restrict__ p) {
  v8bf lo = *(const v8bf*)(p + 0);
  v8bf hi = *(const v8bf*)(p + 16);
  v16bf r;
#pragma unroll
  for (int j = 0; j < 8; ++j) { r[j] = lo[j]; r[8 + j] = hi[j]; }
  return r;
}

// ---------------------------------------------------------------------------
// bf16 WMMA GEMM:  C[M][N] (+ optional bf16 copy) = epi(A[M][K] * BT^T)
//   A  : row-major [M][K], fp32 (converted in-register) or bf16
//   BT : row-major [N][K] bf16 (B transposed, K contiguous per output col)
// Block = 256 threads = 8 waves arranged 2(M) x 4(N); block tile 128x128.
// Wave tile 64x32 = 4x2 WMMA subtiles -> 8 WMMAs share 4 A-frag + 2 B-frag
// loads.  Explicit two-step ping-pong: each iteration loads fragments for
// two future k-steps while issuing the WMMA bursts for the two buffered
// steps -- no loop-carried buffer copies.  KT/32 is even for every call
// site, so the tail is a single compile-time-known step pair.
// amdgpu_waves_per_eu(2) caps the allocation at ~256 VGPRs (no VGPR-MSB
// switching) while keeping 2 waves/SIMD so the cvt VALU work co-executes
// with the other wave's WMMAs.  Dereferences keep the kernarg base pointer
// visible (A + off + k) so address-space inference emits GLOBAL (not FLAT)
// loads.  A is streamed in only N/128 passes; BT stays L2-resident.
// M, N multiples of 128.  EPI: 0 = none, 1 = ReLU.
// ---------------------------------------------------------------------------
template <typename AT, int EPI, bool WBF, int KT>
__global__ __attribute__((amdgpu_waves_per_eu(2)))
__launch_bounds__(256) void wmma_gemm_bf16(
    const AT* __restrict__ A, size_t lda,
    const __bf16* __restrict__ BT, size_t ldb,
    float* __restrict__ C, size_t ldc,
    __bf16* __restrict__ Cbf, size_t ldcbf) {
  static_assert((KT / 32) % 2 == 0 && KT >= 64, "KT/32 must be even");
  const int lane = threadIdx.x;
  const int wv   = threadIdx.y;          // 0..7
  const int half = lane >> 4;            // 0 or 1
  const int l16  = lane & 15;
  const int m0 = blockIdx.y * 128 + (wv & 1) * 64;   // wave M origin
  const int n0 = blockIdx.x * 128 + (wv >> 1) * 32;  // wave N origin

  v8f acc[4][2];
#pragma unroll
  for (int i = 0; i < 4; ++i)
#pragma unroll
    for (int j = 0; j < 2; ++j) acc[i][j] = (v8f)(0.f);

  // Per-fragment row offsets (integer offsets keep loads in GLOBAL space).
  size_t aoff[4], boff[2];
#pragma unroll
  for (int i = 0; i < 4; ++i)
    aoff[i] = (size_t)(m0 + i * 16 + l16) * lda + half * 8;
#pragma unroll
  for (int j = 0; j < 2; ++j)
    boff[j] = (size_t)(n0 + j * 16 + l16) * ldb + half * 8;

  v16bf acur[4], bcur[2], anxt[4], bnxt[2];

  // Prologue: buffer step 0.
#pragma unroll
  for (int i = 0; i < 4; ++i) acur[i] = load_a_frag(A + aoff[i]);
#pragma unroll
  for (int j = 0; j < 2; ++j) bcur[j] = load_b_frag(BT + boff[j]);

  constexpr int S = KT / 32;   // number of 32-wide k-steps (even)

#pragma unroll 1
  for (int s = 1; s + 1 < S; s += 2) {
    const int k0 = s * 32;
    const int k1 = k0 + 32;

    // Load step s while computing step s-1.
#pragma unroll
    for (int i = 0; i < 4; ++i) anxt[i] = load_a_frag(A + aoff[i] + k0);
#pragma unroll
    for (int j = 0; j < 2; ++j) bnxt[j] = load_b_frag(BT + boff[j] + k0);
#pragma unroll
    for (int i = 0; i < 4; ++i)
#pragma unroll
      for (int j = 0; j < 2; ++j)
        acc[i][j] = __builtin_amdgcn_wmma_f32_16x16x32_bf16(
            false, acur[i], false, bcur[j], (short)0, acc[i][j], false, false);

    // Load step s+1 while computing step s.
#pragma unroll
    for (int i = 0; i < 4; ++i) acur[i] = load_a_frag(A + aoff[i] + k1);
#pragma unroll
    for (int j = 0; j < 2; ++j) bcur[j] = load_b_frag(BT + boff[j] + k1);
#pragma unroll
    for (int i = 0; i < 4; ++i)
#pragma unroll
      for (int j = 0; j < 2; ++j)
        acc[i][j] = __builtin_amdgcn_wmma_f32_16x16x32_bf16(
            false, anxt[i], false, bnxt[j], (short)0, acc[i][j], false, false);
  }

  // Tail: step S-1 (S is even, so exactly one buffered step pair remains).
#pragma unroll
  for (int i = 0; i < 4; ++i) anxt[i] = load_a_frag(A + aoff[i] + (S - 1) * 32);
#pragma unroll
  for (int j = 0; j < 2; ++j) bnxt[j] = load_b_frag(BT + boff[j] + (S - 1) * 32);
#pragma unroll
  for (int i = 0; i < 4; ++i)
#pragma unroll
    for (int j = 0; j < 2; ++j)
      acc[i][j] = __builtin_amdgcn_wmma_f32_16x16x32_bf16(
          false, acur[i], false, bcur[j], (short)0, acc[i][j], false, false);
#pragma unroll
  for (int i = 0; i < 4; ++i)
#pragma unroll
    for (int j = 0; j < 2; ++j)
      acc[i][j] = __builtin_amdgcn_wmma_f32_16x16x32_bf16(
          false, anxt[i], false, bnxt[j], (short)0, acc[i][j], false, false);

  // C/D layout: lane owns col n = l16; rows m = half*8 + v (v = VGPR index).
#pragma unroll
  for (int i = 0; i < 4; ++i) {
#pragma unroll
    for (int j = 0; j < 2; ++j) {
      const int n  = n0 + j * 16 + l16;
      const int mb = m0 + i * 16 + half * 8;
#pragma unroll
      for (int v = 0; v < 8; ++v) {
        float x = acc[i][j][v];
        if (EPI == 1) x = x > 0.f ? x : 0.f;
        C[(size_t)(mb + v) * ldc + n] = x;
        if (WBF) Cbf[(size_t)(mb + v) * ldcbf + n] = (__bf16)x;
      }
    }
  }
}

// ---------------------------------------------------------------------------
// Gather + transpose:  outT[f][i] = bf16( src[ idxB?[idxA?[i]] ][ f ] )
// 32x32 tiles via LDS.  n_i, n_f multiples of 32.  blockDim = (32, 8).
// ---------------------------------------------------------------------------
__global__ __launch_bounds__(256) void gather_transpose_bf16(
    const float* __restrict__ src, int ld_src,
    const int* __restrict__ idxA, const int* __restrict__ idxB,
    __bf16* __restrict__ outT, int n_i) {
  __shared__ float tile[32][33];
  const int i0 = blockIdx.x * 32;
  const int f0 = blockIdx.y * 32;

  for (int r = threadIdx.y; r < 32; r += 8) {
    int row = i0 + r;
    if (idxA) row = idxA[row];
    if (idxB) row = idxB[row];
    tile[r][threadIdx.x] = src[(size_t)row * ld_src + f0 + threadIdx.x];
  }
  __syncthreads();
  for (int r = threadIdx.y; r < 32; r += 8) {
    const int f = f0 + r;
    outT[(size_t)f * n_i + i0 + threadIdx.x] = (__bf16)tile[threadIdx.x][r];
  }
}

// ---------------------------------------------------------------------------
// Row gather/convert:  dst[i][col_off + c] = bf16( src[ idxB?[idxA?[i]] ][c] )
// grid.x = n_rows.
// ---------------------------------------------------------------------------
__global__ __launch_bounds__(256) void gather_rows_bf16(
    const float* __restrict__ src, int ld_src,
    const int* __restrict__ idxA, const int* __restrict__ idxB,
    __bf16* __restrict__ dst, int ld_dst, int col_off, int n_cols) {
  int row = blockIdx.x;
  if (idxA) row = idxA[row];
  if (idxB) row = idxB[row];
  const float* s = src + (size_t)row * ld_src;
  __bf16* d = dst + (size_t)blockIdx.x * ld_dst + col_off;
  for (int c = threadIdx.x; c < n_cols; c += blockDim.x) d[c] = (__bf16)s[c];
}

// ---------------------------------------------------------------------------
// Row softmax over 128 columns; one block (128 threads) per row.
// ---------------------------------------------------------------------------
__global__ __launch_bounds__(128) void softmax128(
    const float* __restrict__ logits, float* __restrict__ out) {
  __shared__ float red[128];
  const int r = blockIdx.x, t = threadIdx.x;
  const float x = logits[(size_t)r * 128 + t];

  red[t] = x;
  __syncthreads();
  for (int s = 64; s > 0; s >>= 1) {
    if (t < s) red[t] = fmaxf(red[t], red[t + s]);
    __syncthreads();
  }
  const float m = red[0];
  __syncthreads();

  const float e = __expf(x - m);
  red[t] = e;
  __syncthreads();
  for (int s = 64; s > 0; s >>= 1) {
    if (t < s) red[t] += red[t + s];
    __syncthreads();
  }
  out[(size_t)r * 128 + t] = e / red[0];
}

// ---------------------------------------------------------------------------
// Problem constants (match reference)
// ---------------------------------------------------------------------------
#define Fdim 512
#define Ddim 512
#define Cdim 128
#define S1v  32768
#define B1v  4096
#define S2v  3584
#define B2v  512

static inline size_t align256(size_t x) { return (x + 255) & ~(size_t)255; }

extern "C" void kernel_launch(void* const* d_in, const int* in_sizes, int n_in,
                              void* d_out, int out_size, void* d_ws, size_t ws_size,
                              hipStream_t stream) {
  (void)in_sizes; (void)n_in; (void)out_size; (void)ws_size;
  const float* features = (const float*)d_in[0];
  const int*   src_nodes= (const int*)  d_in[1];
  const int*   dst_idx1 = (const int*)  d_in[2];
  const int*   src_idx1 = (const int*)  d_in[3];
  const float* dif_mat1 = (const float*)d_in[4];
  const int*   dst_idx2 = (const int*)  d_in[5];
  const int*   src_idx2 = (const int*)  d_in[6];
  const float* dif_mat2 = (const float*)d_in[7];
  const float* w1       = (const float*)d_in[8];
  const float* w2       = (const float*)d_in[9];
  const float* w_cls    = (const float*)d_in[10];
  float* out = (float*)d_out;

  // ---- workspace layout (bump allocator, 256B aligned) ----
  char* ws = (char*)d_ws;
  size_t off = 0;
  __bf16* Xs1T = (__bf16*)(ws + off); off = align256(off + (size_t)Fdim * S1v * 2);      // [512][32768]
  __bf16* w1T  = (__bf16*)(ws + off); off = align256(off + (size_t)Ddim * 2 * Fdim * 2); // [512][1024]
  __bf16* w2T  = (__bf16*)(ws + off); off = align256(off + (size_t)Ddim * 2 * Ddim * 2); // [512][1024]
  __bf16* wcT  = (__bf16*)(ws + off); off = align256(off + (size_t)Cdim * Ddim * 2);     // [128][512]
  float*  agg1 = (float*) (ws + off); off = align256(off + (size_t)B1v * Fdim * 4);      // [4096][512]
  __bf16* cat1 = (__bf16*)(ws + off); off = align256(off + (size_t)B1v * 2 * Fdim * 2);  // [4096][1024]
  float*  h1   = (float*) (ws + off); off = align256(off + (size_t)B1v * Ddim * 4);      // [4096][512]
  __bf16* Xs2T = (__bf16*)(ws + off); off = align256(off + (size_t)Ddim * S2v * 2);      // [512][3584]
  float*  agg2 = (float*) (ws + off); off = align256(off + (size_t)B2v * Ddim * 4);      // [512][512]
  __bf16* cat2 = (__bf16*)(ws + off); off = align256(off + (size_t)B2v * 2 * Ddim * 2);  // [512][1024]
  float*  h2f  = (float*) (ws + off); off = align256(off + (size_t)B2v * Ddim * 4);      // [512][512]
  __bf16* h2bf = (__bf16*)(ws + off); off = align256(off + (size_t)B2v * Ddim * 2);      // [512][512]
  float*  logit= (float*) (ws + off); off = align256(off + (size_t)B2v * Cdim * 4);      // [512][128]

  const dim3 tb(32, 8);   // transpose blocks
  const dim3 gb(32, 8);   // GEMM blocks: 8 waves

  // ---- weight transposes (fp32 -> bf16, [K][N] -> [N][K]) ----
  gather_transpose_bf16<<<dim3(2 * Fdim / 32, Ddim / 32), tb, 0, stream>>>(
      w1, Ddim, nullptr, nullptr, w1T, 2 * Fdim);
  gather_transpose_bf16<<<dim3(2 * Ddim / 32, Ddim / 32), tb, 0, stream>>>(
      w2, Ddim, nullptr, nullptr, w2T, 2 * Ddim);
  gather_transpose_bf16<<<dim3(Ddim / 32, Cdim / 32), tb, 0, stream>>>(
      w_cls, Cdim, nullptr, nullptr, wcT, Ddim);

  // ---- layer 1: Xs1T[f][s] = features[src_nodes[src_idx1[s]]][f] ----
  gather_transpose_bf16<<<dim3(S1v / 32, Fdim / 32), tb, 0, stream>>>(
      features, Fdim, src_idx1, src_nodes, Xs1T, S1v);

  // agg1 = dif_mat1 @ Xs1 : M=4096 N=512 K=32768 (the 137-GFLOP GEMM)
  wmma_gemm_bf16<float, 0, false, S1v><<<dim3(Fdim / 128, B1v / 128), gb, 0, stream>>>(
      dif_mat1, S1v, Xs1T, S1v, agg1, Fdim, nullptr, 0);

  // cat1 = [bf16(agg1) | bf16(features[src_nodes[dst_idx1]])]
  gather_rows_bf16<<<B1v, 256, 0, stream>>>(agg1, Fdim, nullptr, nullptr,
                                            cat1, 2 * Fdim, 0, Fdim);
  gather_rows_bf16<<<B1v, 256, 0, stream>>>(features, Fdim, dst_idx1, src_nodes,
                                            cat1, 2 * Fdim, Fdim, Fdim);

  // h1 = relu(cat1 @ w1) : M=4096 N=512 K=1024
  wmma_gemm_bf16<__bf16, 1, false, 2 * Fdim><<<dim3(Ddim / 128, B1v / 128), gb, 0, stream>>>(
      cat1, 2 * Fdim, w1T, 2 * Fdim, h1, Ddim, nullptr, 0);

  // ---- layer 2 ----
  gather_transpose_bf16<<<dim3(S2v / 32, Ddim / 32), tb, 0, stream>>>(
      h1, Ddim, src_idx2, nullptr, Xs2T, S2v);

  // agg2 = dif_mat2 @ h1[src_idx2] : M=512 N=512 K=3584
  wmma_gemm_bf16<float, 0, false, S2v><<<dim3(Ddim / 128, B2v / 128), gb, 0, stream>>>(
      dif_mat2, S2v, Xs2T, S2v, agg2, Ddim, nullptr, 0);

  gather_rows_bf16<<<B2v, 256, 0, stream>>>(agg2, Ddim, nullptr, nullptr,
                                            cat2, 2 * Ddim, 0, Ddim);
  gather_rows_bf16<<<B2v, 256, 0, stream>>>(h1, Ddim, dst_idx2, nullptr,
                                            cat2, 2 * Ddim, Ddim, Ddim);

  // h2 = relu(cat2 @ w2) : M=512 N=512 K=1024  (fp32 + bf16 copies)
  wmma_gemm_bf16<__bf16, 1, true, 2 * Ddim><<<dim3(Ddim / 128, B2v / 128), gb, 0, stream>>>(
      cat2, 2 * Ddim, w2T, 2 * Ddim, h2f, Ddim, h2bf, Ddim);

  // logits = h2 @ w_cls : M=512 N=128 K=512
  wmma_gemm_bf16<__bf16, 0, false, Ddim><<<dim3(Cdim / 128, B2v / 128), gb, 0, stream>>>(
      h2bf, Ddim, wcT, Ddim, logit, Cdim, nullptr, 0);

  // out = softmax(logits, axis=-1)
  softmax128<<<B2v, 128, 0, stream>>>(logit, out);
}
